// GATScheduler_80994493268401
// MI455X (gfx1250) — compile-verified, matrix operand order checked by reading, not run on hardware
//
#include <hip/hip_runtime.h>
#include <hip/hip_bf16.h>

// ---------------------------------------------------------------------------
// GAT scheduler for MI455X (gfx1250, wave32).
// GEMMs: fully-unrolled v_wmma_f32_16x16x32_f16 tiles (K templated),
//        LDS staging with packed f16x2 stores (ds_store_b32) and
//        fragment reads as ds_load_b128 pairs.
// Graph softmax/scatter: float atomics (L2-resident), H/C compile-time.
// ---------------------------------------------------------------------------

#define N_NODES 50000
#define N_EDGES 800000
#define F_NODE  128
#define F_TASK  128
#define HID     64
#define HEADS   4
#define BATCH   16
#define NEG_SLOPE 0.2f

typedef __attribute__((ext_vector_type(16))) _Float16 v16h;
typedef __attribute__((ext_vector_type(8)))  _Float16 v8h;
typedef __attribute__((ext_vector_type(2)))  _Float16 h2f;
typedef __attribute__((ext_vector_type(8)))  float    v8f;

// -------------------------- WMMA GEMM (f32 in, f16 mma, f32 out) -----------
// C[M,N] = A[M,K] @ B[K,N].  One 16x16 D tile per wave, K stepped by 32,
// K known at compile time -> fully unrolled chain of v_wmma.
// REQUIREMENTS (guaranteed by the launcher):
//   M % 16 == 0, gridDim.x == M/16, gridDim.y * (blockDim.x/32) * 16 == N.
// Fragment layouts per CDNA5 ISA 7.12.2 (wave32):
//   A: lane l holds row M=l&15; element i -> K = (i/8)*16 + (l>>4)*8 + i%8.
//   B: lane l holds col N=l&15; same K pattern (Bs staged transposed so both
//      fragments are two contiguous 16-byte LDS reads -> ds_load_b128).
//   D: lane l -> N=l&15, VGPR r -> M = r + (l>>4)*8.
template <int K>
__global__ __launch_bounds__(256)
void gemm_wmma_f16(const float* __restrict__ A, const float* __restrict__ B,
                   float* __restrict__ C, int M, int N) {
  __shared__ _Float16 As[16][32];       // [m][k]
  __shared__ _Float16 Bs[8][16][32];    // [wave][n][k]  (transposed tile)
  const int tid    = threadIdx.x;
  const int nwaves = blockDim.x >> 5;
  const int wave   = tid >> 5;
  const int lane   = tid & 31;
  const int half   = lane >> 4;
  const int l15    = lane & 15;
  const int m0 = blockIdx.x * 16;
  const int n0 = (blockIdx.y * nwaves + wave) * 16;

  v8f acc = {};
  #pragma unroll
  for (int k0 = 0; k0 < K; k0 += 32) {
    // Stage A tile (16 rows x 32 k): float2 global load -> packed f16x2
    // -> single ds_store_b32 per dword (256 dwords total).
    for (int e = tid; e < 256; e += blockDim.x) {
      int r = e >> 4, kp = (e & 15) * 2;
      const float2 av = *(const float2*)&A[(size_t)(m0 + r) * K + k0 + kp];
      *(h2f*)&As[r][kp] = (h2f){(_Float16)av.x, (_Float16)av.y};
    }
    // Stage this wave's B tile (32 k x 16 n) transposed to [n][k].
    // Lane covers column cc=l15, k range [half*16, half*16+16): 8 packed dwords.
    #pragma unroll
    for (int j = 0; j < 8; j++) {
      int cc = l15;
      int kk = half * 16 + j * 2;
      float b0v = B[(size_t)(k0 + kk)     * N + n0 + cc];  // coalesced per row
      float b1v = B[(size_t)(k0 + kk + 1) * N + n0 + cc];
      *(h2f*)&Bs[wave][cc][kk] = (h2f){(_Float16)b0v, (_Float16)b1v};
    }
    __syncthreads();
    // Fragments: two contiguous 8xf16 (16B) runs each -> ds_load_b128 pairs.
    v8h alo = *(const v8h*)&As[l15][half * 8];
    v8h ahi = *(const v8h*)&As[l15][16 + half * 8];
    v8h blo = *(const v8h*)&Bs[wave][l15][half * 8];
    v8h bhi = *(const v8h*)&Bs[wave][l15][16 + half * 8];
    v16h a, b;
    #pragma unroll
    for (int i = 0; i < 8; i++) {
      a[i] = alo[i]; a[8 + i] = ahi[i];
      b[i] = blo[i]; b[8 + i] = bhi[i];
    }
    acc = __builtin_amdgcn_wmma_f32_16x16x32_f16(
        false, a, false, b, (short)0, acc, false, false);
    __syncthreads();
  }
  #pragma unroll
  for (int r = 0; r < 8; r++) {
    int m = m0 + r + (half << 3);
    C[(size_t)m * N + n0 + l15] = acc[r];
  }
}

// -------------------------- attention helpers ------------------------------
__device__ __forceinline__ float lrelu(float x) {
  return x > 0.0f ? x : NEG_SLOPE * x;
}

// Monotone int/uint mapping of IEEE floats -> atomic max without CAS loops.
__device__ __forceinline__ void atomicMaxF(float* addr, float val) {
  if (val >= 0.0f) atomicMax((int*)addr, __float_as_int(val));
  else             atomicMin((unsigned int*)addr, __float_as_uint(val));
}

// alpha_src/alpha_dst: per (node, head) dot of hW row with attention vectors.
template <int H, int C>
__global__ void alpha_kernel(const float* __restrict__ hW,
                             const float* __restrict__ att_src,
                             const float* __restrict__ att_dst,
                             float* __restrict__ as_, float* __restrict__ ad_,
                             int nNodes) {
  int t = blockIdx.x * blockDim.x + threadIdx.x;
  if (t >= nNodes * H) return;
  int h = t % H;                      // H is 1 or 4 -> mask, no divide
  const float* row = hW + (size_t)t * C;
  float s = 0.0f, d = 0.0f;
  #pragma unroll 4
  for (int c = 0; c < C; c++) {
    float v = row[c];
    s += v * att_src[h * C + c];
    d += v * att_dst[h * C + c];
  }
  as_[t] = s;
  ad_[t] = d;
}

// m[n,h] seeded with the self-loop logit (guaranteed member of each segment).
__global__ void init_max_kernel(const float* __restrict__ as_,
                                const float* __restrict__ ad_,
                                float* __restrict__ m, int NH) {
  int t = blockIdx.x * blockDim.x + threadIdx.x;
  if (t < NH) m[t] = lrelu(as_[t] + ad_[t]);
}

template <int H>
__global__ void edge_max_kernel(const int* __restrict__ src, const int* __restrict__ dst,
                                const float* __restrict__ as_, const float* __restrict__ ad_,
                                float* __restrict__ m, int E) {
  int t = blockIdx.x * blockDim.x + threadIdx.x;
  if (t >= E * H) return;
  int e = t / H, h = t % H;           // power-of-two -> shift/mask
  int s = src[e], d = dst[e];
  atomicMaxF(&m[d * H + h], lrelu(as_[s * H + h] + ad_[d * H + h]));
}

__global__ void self_den_kernel(const float* __restrict__ as_,
                                const float* __restrict__ ad_,
                                const float* __restrict__ m,
                                float* __restrict__ den, int NH) {
  int t = blockIdx.x * blockDim.x + threadIdx.x;
  if (t < NH) den[t] = __expf(lrelu(as_[t] + ad_[t]) - m[t]);
}

template <int H>
__global__ void edge_den_kernel(const int* __restrict__ src, const int* __restrict__ dst,
                                const float* __restrict__ as_, const float* __restrict__ ad_,
                                const float* __restrict__ m, float* __restrict__ den,
                                float* __restrict__ exE, int E) {
  int t = blockIdx.x * blockDim.x + threadIdx.x;
  if (t >= E * H) return;
  int e = t / H, h = t % H;
  int s = src[e], d = dst[e];
  float ex = __expf(lrelu(as_[s * H + h] + ad_[d * H + h]) - m[d * H + h]);
  exE[t] = ex;
  atomicAdd(&den[d * H + h], ex);
}

// out[n] = hW[n] * alpha_self  (initializes the accumulator, no atomics)
template <int H, int C>
__global__ void self_agg_kernel(const float* __restrict__ hW,
                                const float* __restrict__ as_, const float* __restrict__ ad_,
                                const float* __restrict__ m, const float* __restrict__ den,
                                float* __restrict__ out, int nNodes) {
  int t = blockIdx.x * blockDim.x + threadIdx.x;
  if (t >= nNodes * H * C) return;
  int nh = t / C;                     // (n,h) flat index; C=64 -> shift
  float a = __expf(lrelu(as_[nh] + ad_[nh]) - m[nh]) / (den[nh] + 1e-16f);
  out[t] = hW[t] * a;
}

// One block per edge; blockDim == H*C. out[dst] += hW[src] * alpha(edge).
template <int H, int C>
__global__ void edge_agg_kernel(const int* __restrict__ src, const int* __restrict__ dst,
                                const float* __restrict__ hW, const float* __restrict__ exE,
                                const float* __restrict__ den, float* __restrict__ out) {
  int e = blockIdx.x;
  int idx = threadIdx.x;              // h*C + c
  int h = idx / C;                    // shift
  int s = src[e], d = dst[e];
  float alpha = exE[(size_t)e * H + h] / (den[d * H + h] + 1e-16f);
  atomicAdd(&out[(size_t)d * H * C + idx], hW[(size_t)s * H * C + idx] * alpha);
}

template <int HC>
__global__ void finalize_elu_kernel(float* __restrict__ out,
                                    const float* __restrict__ bias, int total) {
  int t = blockIdx.x * blockDim.x + threadIdx.x;
  if (t >= total) return;
  float v = out[t] + bias[t & (HC - 1)];   // HC is 256 or 64
  out[t] = v > 0.0f ? v : (__expf(v) - 1.0f);
}

// -------------------------- task encoder + t@Wt ----------------------------
// One block per batch row; 64 threads (one per hidden channel).
__global__ void task_kernel(const float* __restrict__ task_feat,
                            const float* __restrict__ te_W1, const float* __restrict__ te_b1,
                            const float* __restrict__ te_W2, const float* __restrict__ te_b2,
                            const float* __restrict__ Wt,   // cl_W1 + HID*HID
                            float* __restrict__ tWt) {
  __shared__ float hid[HID];
  __shared__ float t2[HID];
  int b = blockIdx.x, c = threadIdx.x;
  float acc = te_b1[c];
  for (int k = 0; k < F_TASK; k++) acc += task_feat[b * F_TASK + k] * te_W1[k * HID + c];
  hid[c] = acc > 0.0f ? acc : 0.0f;
  __syncthreads();
  float acc2 = te_b2[c];
  for (int k = 0; k < HID; k++) acc2 += hid[k] * te_W2[k * HID + c];
  t2[c] = acc2;
  __syncthreads();
  float acc3 = 0.0f;
  for (int k = 0; k < HID; k++) acc3 += t2[k] * Wt[k * HID + c];
  tWt[b * HID + c] = acc3;
}

// -------------------------- classifier -------------------------------------
// One wave32 per node; lane owns channels (l, l+32); shuffle-reduce per batch.
__global__ __launch_bounds__(256)
void classifier_kernel(const float* __restrict__ hWh, const float* __restrict__ tWt,
                       const float* __restrict__ cl_b1, const float* __restrict__ cl_W2,
                       const float* __restrict__ cl_b2, float* __restrict__ scores,
                       int nNodes) {
  int wave = threadIdx.x >> 5;
  int lane = threadIdx.x & 31;
  int n = blockIdx.x * 8 + wave;
  if (n >= nNodes) return;
  float hv0 = hWh[(size_t)n * HID + lane];
  float hv1 = hWh[(size_t)n * HID + 32 + lane];
  float b0 = cl_b1[lane], b1 = cl_b1[32 + lane];
  float w0 = cl_W2[lane], w1 = cl_W2[32 + lane];
  float bias2 = cl_b2[0];
  #pragma unroll
  for (int b = 0; b < BATCH; b++) {
    float z0 = hv0 + tWt[b * HID + lane] + b0;      z0 = z0 > 0.0f ? z0 : 0.0f;
    float z1 = hv1 + tWt[b * HID + 32 + lane] + b1; z1 = z1 > 0.0f ? z1 : 0.0f;
    float p = z0 * w0 + z1 * w1;
    #pragma unroll
    for (int off = 16; off > 0; off >>= 1) p += __shfl_xor(p, off, 32);
    if (lane == 0) scores[(size_t)b * nNodes + n] = p + bias2;
  }
}

// -------------------------- launcher ---------------------------------------
extern "C" void kernel_launch(void* const* d_in, const int* in_sizes, int n_in,
                              void* d_out, int out_size, void* d_ws, size_t ws_size,
                              hipStream_t stream) {
  const float* x        = (const float*)d_in[0];
  const int*   ei       = (const int*)  d_in[1];
  const float* task     = (const float*)d_in[2];
  const float* W1       = (const float*)d_in[3];
  const float* att_s1   = (const float*)d_in[4];
  const float* att_d1   = (const float*)d_in[5];
  const float* b1       = (const float*)d_in[6];
  const float* W2       = (const float*)d_in[7];
  const float* att_s2   = (const float*)d_in[8];
  const float* att_d2   = (const float*)d_in[9];
  const float* b2       = (const float*)d_in[10];
  const float* te_W1    = (const float*)d_in[11];
  const float* te_b1    = (const float*)d_in[12];
  const float* te_W2    = (const float*)d_in[13];
  const float* te_b2    = (const float*)d_in[14];
  const float* cl_W1    = (const float*)d_in[15];
  const float* cl_b1    = (const float*)d_in[16];
  const float* cl_W2    = (const float*)d_in[17];
  const float* cl_b2    = (const float*)d_in[18];
  const int* src = ei;
  const int* dst = ei + N_EDGES;

  // Workspace carving (floats).
  float* ws = (float*)d_ws;
  const size_t NN = N_NODES, NE = N_EDGES;
  float* hW1  = ws;                    // NN*256
  float* out1 = hW1  + NN * 256;       // NN*256  -> becomes h1 after ELU
  float* hW2  = out1 + NN * 256;       // NN*64
  float* out2 = hW2  + NN * 64;        // NN*64   -> becomes h2 after ELU
  float* hWh  = out2 + NN * 64;        // NN*64
  float* as1  = hWh  + NN * 64;        // NN*4
  float* ad1  = as1  + NN * 4;
  float* m1   = ad1  + NN * 4;
  float* den1 = m1   + NN * 4;
  float* as2  = den1 + NN * 4;         // NN
  float* ad2  = as2  + NN;
  float* m2   = ad2  + NN;
  float* den2 = m2   + NN;
  float* exE1 = den2 + NN;             // NE*4
  float* exE2 = exE1 + NE * 4;         // NE
  float* tWt  = exE2 + NE;             // BATCH*HID

  const int TB = 256;
  auto cdiv = [](int a, int b) { return (a + b - 1) / b; };

  // ---- GAT layer 1 ----
  // N=256: 8 waves/block x 16 cols x gridDim.y=2 == 256. M/16 = 3125 exactly.
  gemm_wmma_f16<F_NODE><<<dim3(N_NODES / 16, 2), 256, 0, stream>>>(
      x, W1, hW1, N_NODES, HEADS * HID);
  alpha_kernel<HEADS, HID><<<cdiv(N_NODES * HEADS, TB), TB, 0, stream>>>(
      hW1, att_s1, att_d1, as1, ad1, N_NODES);
  init_max_kernel<<<cdiv(N_NODES * HEADS, TB), TB, 0, stream>>>(as1, ad1, m1,
                                                                N_NODES * HEADS);
  edge_max_kernel<HEADS><<<cdiv(N_EDGES * HEADS, TB), TB, 0, stream>>>(
      src, dst, as1, ad1, m1, N_EDGES);
  self_den_kernel<<<cdiv(N_NODES * HEADS, TB), TB, 0, stream>>>(as1, ad1, m1, den1,
                                                                N_NODES * HEADS);
  edge_den_kernel<HEADS><<<cdiv(N_EDGES * HEADS, TB), TB, 0, stream>>>(
      src, dst, as1, ad1, m1, den1, exE1, N_EDGES);
  self_agg_kernel<HEADS, HID><<<cdiv(N_NODES * HEADS * HID, TB), TB, 0, stream>>>(
      hW1, as1, ad1, m1, den1, out1, N_NODES);
  edge_agg_kernel<HEADS, HID><<<N_EDGES, HEADS * HID, 0, stream>>>(
      src, dst, hW1, exE1, den1, out1);
  finalize_elu_kernel<HEADS * HID><<<cdiv(N_NODES * HEADS * HID, TB), TB, 0, stream>>>(
      out1, b1, N_NODES * HEADS * HID);                // out1 == h1

  // ---- GAT layer 2 (heads=1) ----
  // N=64: 4 waves/block x 16 cols x gridDim.y=1 == 64.
  gemm_wmma_f16<HEADS * HID><<<dim3(N_NODES / 16, 1), 128, 0, stream>>>(
      out1, W2, hW2, N_NODES, HID);
  alpha_kernel<1, HID><<<cdiv(N_NODES, TB), TB, 0, stream>>>(
      hW2, att_s2, att_d2, as2, ad2, N_NODES);
  init_max_kernel<<<cdiv(N_NODES, TB), TB, 0, stream>>>(as2, ad2, m2, N_NODES);
  edge_max_kernel<1><<<cdiv(N_EDGES, TB), TB, 0, stream>>>(src, dst, as2, ad2, m2,
                                                           N_EDGES);
  self_den_kernel<<<cdiv(N_NODES, TB), TB, 0, stream>>>(as2, ad2, m2, den2, N_NODES);
  edge_den_kernel<1><<<cdiv(N_EDGES, TB), TB, 0, stream>>>(src, dst, as2, ad2, m2,
                                                           den2, exE2, N_EDGES);
  self_agg_kernel<1, HID><<<cdiv(N_NODES * HID, TB), TB, 0, stream>>>(
      hW2, as2, ad2, m2, den2, out2, N_NODES);
  edge_agg_kernel<1, HID><<<N_EDGES, HID, 0, stream>>>(src, dst, hW2, exE2, den2, out2);
  finalize_elu_kernel<HID><<<cdiv(N_NODES * HID, TB), TB, 0, stream>>>(
      out2, b2, N_NODES * HID);                        // out2 == h2

  // ---- classifier node-side GEMM: hWh = h2 @ cl_W1[:HID] ----
  gemm_wmma_f16<HID><<<dim3(N_NODES / 16, 1), 128, 0, stream>>>(
      out2, cl_W1, hWh, N_NODES, HID);

  // ---- task encoder fused with t @ cl_W1[HID:] ----
  task_kernel<<<BATCH, HID, 0, stream>>>(task, te_W1, te_b1, te_W2, te_b2,
                                         cl_W1 + HID * HID, tWt);

  // ---- final scores ----
  classifier_kernel<<<cdiv(N_NODES, 8), TB, 0, stream>>>(
      hWh, tWt, cl_b1, cl_W2, cl_b2, (float*)d_out, N_NODES);
}
